// SelfAttentionWithResidual_50672024158517
// MI455X (gfx1250) — matrix operand order repaired and checked by reading, not compiled, not run on hardware
//
#include <hip/hip_runtime.h>

typedef __attribute__((ext_vector_type(16))) _Float16 v16h;
typedef __attribute__((ext_vector_type(8)))  float    v8f;

#define BATCH 128
#define DIM   1024
#define LN_EPS 1e-3f
#define LOG2E 1.44269504088896340736f

__global__ __launch_bounds__(256)
void attn_flash_wmma_kernel(const float* __restrict__ x,
                            const float* __restrict__ wq,
                            const float* __restrict__ wk,
                            const float* __restrict__ wv,
                            const float* __restrict__ gamma,
                            const float* __restrict__ beta,
                            float* __restrict__ out) {
    __shared__ float xs[DIM];      // x row (f32)
    __shared__ float numer[DIM];   // sum_j e_ij * x_j
    __shared__ float denom[DIM];   // sum_j e_ij
    __shared__ float red[16];
    __shared__ float sc[2];

    const int b    = blockIdx.x;
    const int tid  = threadIdx.x;
    const int lane = tid & 31;     // wave32
    const int wave = tid >> 5;     // 8 waves
    const float* __restrict__ xb = x + b * DIM;

    // ---- stage x row into LDS (coalesced b128) ----
    float4* xs4 = (float4*)xs;
    const float4 v4 = ((const float4*)xb)[tid];
    xs4[tid] = v4;
    __syncthreads();

    // ---- row max/min (exact softmax max: scores affine in x_j) ----
    float mx = fmaxf(fmaxf(v4.x, v4.y), fmaxf(v4.z, v4.w));
    float mn = fminf(fminf(v4.x, v4.y), fminf(v4.z, v4.w));
    #pragma unroll
    for (int off = 16; off > 0; off >>= 1) {
        mx = fmaxf(mx, __shfl_down(mx, off, 32));
        mn = fminf(mn, __shfl_down(mn, off, 32));
    }
    if (lane == 0) { red[wave] = mx; red[8 + wave] = mn; }
    __syncthreads();
    if (tid == 0) {
        float gmx = red[0], gmn = red[8];
        #pragma unroll
        for (int w = 1; w < 8; ++w) { gmx = fmaxf(gmx, red[w]); gmn = fminf(gmn, red[8 + w]); }
        sc[0] = gmx; sc[1] = gmn;
    }
    __syncthreads();
    const float xmax = sc[0], xmin = sc[1];
    // fold log2(e) into the per-row affine coefficients so the hot loop is
    // a bare v_fma_f32 + v_exp_f32 (exp2), no per-element ln2 scaling mul
    const float cscl = wq[0] * wk[0] * (LOG2E / (float)DIM);
    const float sv   = wv[0];

    // ---- flash-attention core: A = exp tile (16i x 32j), B col0 = x_j, col1 = 1 ----
    const int n_ = lane & 15;            // A: row within tile / B: column / D: column
    const int hi = lane >> 4;            // half-wave selects K sub-range
    const int wavebase = wave * 128;     // 128 rows per wave -> 8 tiles of 16

    float a_t[8], m_t[8];
    #pragma unroll
    for (int t = 0; t < 8; ++t) {
        const float a = cscl * xs[wavebase + t * 16 + n_];   // a_i * log2e
        a_t[t] = a;
        m_t[t] = fmaxf(a * xmax, a * xmin);                  // m_i * log2e
    }

    v8f acc[8];
    #pragma unroll
    for (int t = 0; t < 8; ++t)
        #pragma unroll
        for (int e = 0; e < 8; ++e) acc[t][e] = 0.0f;

    for (int jb = 0; jb < 32; ++jb) {
        const int j0 = jb * 32;

        // x values in the A-fragment K pattern: kA(v,p) = 16*(v/4) + 8*hi + 2*(v&3) + p
        float xa[16];
        #pragma unroll
        for (int v = 0; v < 8; ++v) {
            const int k = 16 * (v >> 2) + 8 * hi + 2 * (v & 3);
            const float2 p2 = *(const float2*)&xs[j0 + k];   // lane-uniform -> LDS broadcast
            xa[2 * v]     = p2.x;
            xa[2 * v + 1] = p2.y;
        }

        // B fragment: K = 16*(v/4) + 8*hi + 2*(v&3) + p, column n_: col0 = x_j, col1 = 1, rest 0
        v16h bfrag;
        #pragma unroll
        for (int v = 0; v < 8; ++v) {
            const int k = 16 * (v >> 2) + 8 * hi + 2 * (v & 3);
            const float2 p2 = *(const float2*)&xs[j0 + k];
            _Float16 b0, b1;
            if (n_ == 0)      { b0 = (_Float16)p2.x; b1 = (_Float16)p2.y; }
            else if (n_ == 1) { b0 = (_Float16)1.0f; b1 = (_Float16)1.0f; }
            else              { b0 = (_Float16)0.0f; b1 = (_Float16)0.0f; }
            bfrag[2 * v]     = b0;
            bfrag[2 * v + 1] = b1;
        }

        // 8 row-tiles reuse the same x loads and B fragment
        #pragma unroll
        for (int t = 0; t < 8; ++t) {
            const float a = a_t[t], mi = m_t[t];
            v16h afrag;
            #pragma unroll
            for (int e = 0; e < 16; ++e)
                afrag[e] = (_Float16)__builtin_amdgcn_exp2f(__builtin_fmaf(a, xa[e], -mi));
            acc[t] = __builtin_amdgcn_wmma_f32_16x16x32_f16(
                false, afrag, false, bfrag, (short)0, acc[t], false, false);
        }
    }

    // ---- D layout: D[m,n] -> VGPR (m&7), lane n + 16*(m>>3). numer = col0, denom = col1 ----
    #pragma unroll
    for (int t = 0; t < 8; ++t) {
        if (n_ < 2) {
            float* dst = (n_ == 0) ? numer : denom;
            const int base = wavebase + t * 16 + hi * 8;
            #pragma unroll
            for (int v = 0; v < 8; ++v) dst[base + v] = acc[t][v];
        }
    }
    __syncthreads();

    // ---- residual + LayerNorm ----
    float yv[4], s1 = 0.0f, s2 = 0.0f;
    #pragma unroll
    for (int k = 0; k < 4; ++k) {
        const int i = tid + k * 256;
        const float y = xs[i] + sv * numer[i] / denom[i];
        yv[k] = y;
        s1 += y;
        s2 += y * y;
    }
    #pragma unroll
    for (int off = 16; off > 0; off >>= 1) {
        s1 += __shfl_down(s1, off, 32);
        s2 += __shfl_down(s2, off, 32);
    }
    if (lane == 0) { red[wave] = s1; red[8 + wave] = s2; }
    __syncthreads();
    if (tid == 0) {
        float t1 = 0.0f, t2 = 0.0f;
        #pragma unroll
        for (int w = 0; w < 8; ++w) { t1 += red[w]; t2 += red[8 + w]; }
        const float mean = t1 * (1.0f / (float)DIM);
        const float var  = t2 * (1.0f / (float)DIM) - mean * mean;
        sc[0] = mean;
        sc[1] = __frsqrt_rn(var + LN_EPS);
    }
    __syncthreads();
    const float mean = sc[0], rstd = sc[1];

    float* __restrict__ ob = out + b * DIM;
    #pragma unroll
    for (int k = 0; k < 4; ++k) {
        const int i = tid + k * 256;
        ob[i] = (yv[k] - mean) * rstd * gamma[i] + beta[i];
    }
}

extern "C" void kernel_launch(void* const* d_in, const int* in_sizes, int n_in,
                              void* d_out, int out_size, void* d_ws, size_t ws_size,
                              hipStream_t stream) {
    const float* x     = (const float*)d_in[0];
    const float* wq    = (const float*)d_in[1];
    const float* wk    = (const float*)d_in[2];
    const float* wv    = (const float*)d_in[3];
    const float* gamma = (const float*)d_in[4];
    const float* beta  = (const float*)d_in[5];
    float* out = (float*)d_out;

    hipLaunchKernelGGL(attn_flash_wmma_kernel, dim3(BATCH), dim3(256), 0, stream,
                       x, wq, wk, wv, gamma, beta, out);
}
